// MultiHeadTemporalAttention_91182155694642
// MI455X (gfx1250) — matrix operand order repaired and checked by reading, not compiled, not run on hardware
//
#include <hip/hip_runtime.h>
#include <hip/hip_bf16.h>

// ---------------------------------------------------------------------------
// Fused transformer block for MI455X (gfx1250, wave32, WMMA bf16).
//   residual + Wout( softmax(Q K^T / sqrt(hd)) V ) with LN + QKV fused in.
// All matmuls on v_wmma_f32_16x16x32_bf16; GEMM staging via async-to-LDS
// (GLOBAL_LOAD_ASYNC_TO_LDS_B128 + ASYNCcnt) with LDS double buffering.
// ---------------------------------------------------------------------------

#define D_MODEL   1024
#define N_HEADS   16
#define HEAD_DIM  64
#define BATCH     2
#define SEQ       2048
#define BT        (BATCH * SEQ)      // 4096 tokens
#define D3        (3 * D_MODEL)      // 3072
#define LDSK      40                 // padded K stride in LDS (bf16 elems)

typedef __attribute__((ext_vector_type(16))) __bf16       v16bf;
typedef __attribute__((ext_vector_type(8)))  float        v8f;
typedef __attribute__((ext_vector_type(4)))  unsigned int u32x4;
typedef __attribute__((ext_vector_type(4)))  int          v4i;

#if defined(__AMDGCN__) && \
    __has_builtin(__builtin_amdgcn_global_load_async_to_lds_b128) && \
    __has_builtin(__builtin_amdgcn_s_wait_asynccnt)
#define HAVE_ASYNC_LDS 1
#else
#define HAVE_ASYNC_LDS 0
#endif

static __device__ __forceinline__ void async_copy16(void* lds_dst, const void* gsrc) {
#if HAVE_ASYNC_LDS
  __builtin_amdgcn_global_load_async_to_lds_b128(
      (__attribute__((address_space(1))) v4i*)gsrc,
      (__attribute__((address_space(3))) v4i*)lds_dst,
      /*offset=*/0, /*cpol=*/0);
#else
  *(u32x4*)lds_dst = *(const u32x4*)gsrc;
#endif
}

static __device__ __forceinline__ void wait_async_all() {
#if HAVE_ASYNC_LDS
  __builtin_amdgcn_s_wait_asynccnt(0);
#endif
}

static __device__ __forceinline__ unsigned short f2bf(float f) {
  unsigned int u = __float_as_uint(f);
  return (unsigned short)((u + 0x7FFFu + ((u >> 16) & 1u)) >> 16);
}

// A fragment (16x32 bf16, M x K): lane = row (m = lane&15).
// lanes 0-15 : u0 = k[0..7],  u1 = k[16..23]
// lanes 16-31: u0 = k[8..15], u1 = k[24..31]
static __device__ __forceinline__ v16bf load_frag_a(const unsigned short* src, int ld) {
  const int lane = threadIdx.x & 31;
  const int m    = lane & 15;
  const int off  = (lane >> 4) << 3;                 // 0 or 8
  union { v16bf v; u32x4 u[2]; } f;
  f.u[0] = *(const u32x4*)(src + (size_t)m * ld + off);
  f.u[1] = *(const u32x4*)(src + (size_t)m * ld + off + 16);
  return f.v;
}

// B fragment (32x16 bf16, K x N) from a [N][K] (k-contiguous) source:
// lane = col (n = lane&15); lanes 0-15 k=0..7,16..23, lanes 16-31 k=8..15,24..31.
static __device__ __forceinline__ v16bf load_frag_b(const unsigned short* src, int ld) {
  const int lane = threadIdx.x & 31;
  const int n    = lane & 15;
  const int off  = (lane >> 4) << 3;                 // 0 or 8
  union { v16bf v; u32x4 u[2]; } f;
  f.u[0] = *(const u32x4*)(src + (size_t)n * ld + off);
  f.u[1] = *(const u32x4*)(src + (size_t)n * ld + off + 16);
  return f.v;
}

static __device__ __forceinline__ v8f wmma_bf16(v16bf a, v16bf b, v8f c) {
  return __builtin_amdgcn_wmma_f32_16x16x32_bf16(
      /*neg_a=*/false, a, /*neg_b=*/false, b,
      /*c_mod=*/(short)0, c, /*reuse_a=*/false, /*reuse_b=*/false);
}

// ---------------------------------------------------------------------------
// Kernel 1: LayerNorm(x) -> bf16 h   (one block per token row)
// ---------------------------------------------------------------------------
__global__ __launch_bounds__(256) void ln_bf16_kernel(
    const float* __restrict__ x, const float* __restrict__ gamma,
    const float* __restrict__ beta, unsigned short* __restrict__ h) {
  __shared__ float2 red[256];
  const int tid = threadIdx.x;
  const size_t row = blockIdx.x;
  const float* xr = x + row * D_MODEL;

  float vals[4];
  float s = 0.f, s2 = 0.f;
  for (int i = 0; i < 4; ++i) {
    float v = xr[tid + 256 * i];
    vals[i] = v; s += v; s2 += v * v;
  }
  red[tid] = make_float2(s, s2);
  __syncthreads();
  for (int off = 128; off > 0; off >>= 1) {
    if (tid < off) {
      red[tid].x += red[tid + off].x;
      red[tid].y += red[tid + off].y;
    }
    __syncthreads();
  }
  const float mean = red[0].x * (1.0f / D_MODEL);
  const float var  = red[0].y * (1.0f / D_MODEL) - mean * mean;
  const float rstd = rsqrtf(var + 1e-5f);
  for (int i = 0; i < 4; ++i) {
    int c = tid + 256 * i;
    float v = (vals[i] - mean) * rstd * gamma[c] + beta[c];
    h[row * D_MODEL + c] = f2bf(v);
  }
}

// ---------------------------------------------------------------------------
// Kernel 2: W[K][N] f32 -> Wt[N][K] bf16 (B-operand wants k-contiguous cols)
// ---------------------------------------------------------------------------
__global__ __launch_bounds__(256) void transpose_f32_to_bf16(
    const float* __restrict__ src, unsigned short* __restrict__ dst,
    int K, int N) {
  long long i = (long long)blockIdx.x * 256 + threadIdx.x;
  long long total = (long long)K * N;
  if (i >= total) return;
  int n  = (int)(i / K);
  int kk = (int)(i % K);
  dst[i] = f2bf(src[(long long)kk * N + n]);
}

// ---------------------------------------------------------------------------
// Shared GEMM core: 128x128 block tile, 8 waves, each wave 64x32 (4x2 WMMA),
// K-step 32, async double-buffered LDS staging.  A is [M][K] bf16 row-major,
// B is [N][K] bf16 (k-contiguous).  Accumulators handed to caller epilogue.
// ---------------------------------------------------------------------------
struct GemmAcc { v8f acc[4][2]; };

static __device__ __forceinline__ void gemm_core_128x128(
    const unsigned short* __restrict__ hA,   // block's A base: [128][D_MODEL]
    const unsigned short* __restrict__ wB,   // block's B base: [128][D_MODEL]
    unsigned short (*As)[128][LDSK],         // [2][128][LDSK]
    unsigned short (*Bs)[128][LDSK],
    GemmAcc& g) {
  const int tid  = threadIdx.x;
  const int wave = tid >> 5;
  const int wm   = wave >> 2;          // 0..1  (64 rows each)
  const int wn   = wave & 3;           // 0..3  (32 cols each)

  // Stage one 128x32 bf16 tile (512 x 16B chunks, 2 per thread per matrix).
  auto stage = [&](int buf, int kb) {
    for (int idx = tid; idx < 512; idx += 256) {
      const int r = idx >> 2, c = (idx & 3) << 3;
      async_copy16(&As[buf][r][c], hA + (size_t)r * D_MODEL + kb + c);
      async_copy16(&Bs[buf][r][c], wB + (size_t)r * D_MODEL + kb + c);
    }
  };

  stage(0, 0);
  int buf = 0;
  for (int kb = 0; kb < D_MODEL; kb += 32) {
    wait_async_all();          // this wave's async writes to LDS landed
    __syncthreads();           // everyone's writes landed / prior reads done
    if (kb + 32 < D_MODEL) stage(buf ^ 1, kb + 32);

    v16bf af[4], bfr[2];
    for (int i = 0; i < 4; ++i)
      af[i] = load_frag_a(&As[buf][wm * 64 + i * 16][0], LDSK);
    for (int j = 0; j < 2; ++j)
      bfr[j] = load_frag_b(&Bs[buf][wn * 32 + j * 16][0], LDSK);
    for (int i = 0; i < 4; ++i)
      for (int j = 0; j < 2; ++j)
        g.acc[i][j] = wmma_bf16(af[i], bfr[j], g.acc[i][j]);

    buf ^= 1;
  }
}

// ---------------------------------------------------------------------------
// Kernel 3: QKV GEMM  h[BT][D] x Wqkv -> q,k (bf16 [B,H,T,hd]) and vT
// (bf16 [B,H,hd,T]).  Q pre-scaled by 1/sqrt(hd).
// ---------------------------------------------------------------------------
__global__ __launch_bounds__(256) void qkv_gemm_kernel(
    const unsigned short* __restrict__ h, const unsigned short* __restrict__ wT,
    const float* __restrict__ bias,
    unsigned short* __restrict__ q, unsigned short* __restrict__ k,
    unsigned short* __restrict__ vT) {
  __shared__ __align__(16) unsigned short As[2][128][LDSK];
  __shared__ __align__(16) unsigned short Bs[2][128][LDSK];

  const int lane = threadIdx.x & 31;
  const int wave = threadIdx.x >> 5;
  const int wm   = wave >> 2, wn = wave & 3;
  const int tileM = blockIdx.y;        // 32 tiles of 128 rows
  const int tileN = blockIdx.x;        // 24 tiles of 128 cols

  GemmAcc g = {};
  gemm_core_128x128(h  + (size_t)tileM * 128 * D_MODEL,
                    wT + (size_t)tileN * 128 * D_MODEL, As, Bs, g);

  const int hi = lane >> 4, ln = lane & 15;
  for (int i = 0; i < 4; ++i) {
    for (int j = 0; j < 2; ++j) {
      const int c = tileN * 128 + wn * 32 + j * 16 + ln;
      const float bv = bias[c];
      const int part = c >> 10;               // 0:q 1:k 2:v
      const int d    = c & 1023;
      const int head = d >> 6, hdi = d & 63;
      for (int a = 0; a < 8; ++a) {
        const int r = tileM * 128 + wm * 64 + i * 16 + hi * 8 + a;
        const int b = r >> 11, t = r & 2047;
        const size_t bh = (size_t)(b * N_HEADS + head);
        const float val = g.acc[i][j][a] + bv;
        if (part == 0)
          q[(bh * SEQ + t) * HEAD_DIM + hdi] = f2bf(val * 0.125f);
        else if (part == 1)
          k[(bh * SEQ + t) * HEAD_DIM + hdi] = f2bf(val);
        else
          vT[(bh * HEAD_DIM + hdi) * SEQ + t] = f2bf(val);
      }
    }
  }
}

// ---------------------------------------------------------------------------
// Kernel 4: flash attention.  Workgroup = 4 waves = 64 query rows of one
// (b,h).  Q fragments persistent in VGPRs; stream key tiles of 32 with
// online softmax; P re-layouts through LDS into the A-operand of P.V.
// ---------------------------------------------------------------------------
__global__ __launch_bounds__(128) void attention_kernel(
    const unsigned short* __restrict__ q, const unsigned short* __restrict__ k,
    const unsigned short* __restrict__ vT, unsigned short* __restrict__ attn) {
  __shared__ __align__(16) unsigned short Pl[4][16][LDSK];   // per-wave P tile

  const int lane = threadIdx.x & 31;
  const int wave = threadIdx.x >> 5;
  const int hi   = lane >> 4, ln = lane & 15;
  const int bh    = blockIdx.y;                 // 0..31
  const int qBase = blockIdx.x * 64 + wave * 16;

  const unsigned short* qptr = q + ((size_t)bh * SEQ + qBase) * HEAD_DIM;
  const v16bf qa0 = load_frag_a(qptr,      HEAD_DIM);   // d 0..31
  const v16bf qa1 = load_frag_a(qptr + 32, HEAD_DIM);   // d 32..63

  v8f O[4] = {};
  float rmax[8], rsum[8];
  for (int a = 0; a < 8; ++a) { rmax[a] = -3.0e38f; rsum[a] = 0.f; }

  for (int kt = 0; kt < SEQ; kt += 32) {
    const unsigned short* kptr = k + ((size_t)bh * SEQ + kt) * HEAD_DIM;
    if (kt + 32 < SEQ) {                      // prefetch next K tile
      __builtin_prefetch(kptr + 32 * HEAD_DIM + (size_t)lane * 64, 0, 1);
    }
    v8f s0 = {}, s1 = {};
    s0 = wmma_bf16(qa0, load_frag_b(kptr, HEAD_DIM), s0);
    s0 = wmma_bf16(qa1, load_frag_b(kptr + 32, HEAD_DIM), s0);
    s1 = wmma_bf16(qa0, load_frag_b(kptr + 16 * HEAD_DIM, HEAD_DIM), s1);
    s1 = wmma_bf16(qa1, load_frag_b(kptr + 16 * HEAD_DIM + 32, HEAD_DIM), s1);

    float p0[8], p1[8];
    for (int a = 0; a < 8; ++a) {
      float tm = fmaxf(s0[a], s1[a]);                  // row = a + 8*hi
      tm = fmaxf(tm, __shfl_xor(tm, 1, 16));
      tm = fmaxf(tm, __shfl_xor(tm, 2, 16));
      tm = fmaxf(tm, __shfl_xor(tm, 4, 16));
      tm = fmaxf(tm, __shfl_xor(tm, 8, 16));
      const float nm   = fmaxf(rmax[a], tm);
      const float corr = __expf(rmax[a] - nm);
      rmax[a] = nm;
      p0[a] = __expf(s0[a] - nm);
      p1[a] = __expf(s1[a] - nm);
      float ps = p0[a] + p1[a];
      ps += __shfl_xor(ps, 1, 16);
      ps += __shfl_xor(ps, 2, 16);
      ps += __shfl_xor(ps, 4, 16);
      ps += __shfl_xor(ps, 8, 16);
      rsum[a] = rsum[a] * corr + ps;
      O[0][a] *= corr; O[1][a] *= corr; O[2][a] *= corr; O[3][a] *= corr;
    }

    __syncthreads();
    for (int a = 0; a < 8; ++a) {
      Pl[wave][hi * 8 + a][ln]      = f2bf(p0[a]);
      Pl[wave][hi * 8 + a][16 + ln] = f2bf(p1[a]);
    }
    __syncthreads();

    const v16bf pa = load_frag_a(&Pl[wave][0][0], LDSK);
    const unsigned short* vp = vT + (size_t)bh * HEAD_DIM * SEQ + kt;
    for (int oc = 0; oc < 4; ++oc)
      O[oc] = wmma_bf16(pa, load_frag_b(vp + (size_t)oc * 16 * SEQ, SEQ), O[oc]);
  }

  const int b = bh >> 4, hh = bh & 15;
  for (int a = 0; a < 8; ++a) {
    const float inv = 1.0f / rsum[a];
    const int t = qBase + hi * 8 + a;
    const size_t row = (size_t)b * SEQ + t;
    for (int oc = 0; oc < 4; ++oc) {
      const int col = hh * HEAD_DIM + oc * 16 + ln;
      attn[row * D_MODEL + col] = f2bf(O[oc][a] * inv);
    }
  }
}

// ---------------------------------------------------------------------------
// Kernel 5: out projection + bias + residual, f32 output.
// ---------------------------------------------------------------------------
__global__ __launch_bounds__(256) void out_gemm_kernel(
    const unsigned short* __restrict__ A, const unsigned short* __restrict__ wT,
    const float* __restrict__ bias, const float* __restrict__ resid,
    float* __restrict__ out) {
  __shared__ __align__(16) unsigned short As[2][128][LDSK];
  __shared__ __align__(16) unsigned short Bs[2][128][LDSK];

  const int lane = threadIdx.x & 31;
  const int wave = threadIdx.x >> 5;
  const int wm   = wave >> 2, wn = wave & 3;
  const int tileM = blockIdx.y;
  const int tileN = blockIdx.x;

  GemmAcc g = {};
  gemm_core_128x128(A  + (size_t)tileM * 128 * D_MODEL,
                    wT + (size_t)tileN * 128 * D_MODEL, As, Bs, g);

  const int hi = lane >> 4, ln = lane & 15;
  for (int i = 0; i < 4; ++i) {
    for (int j = 0; j < 2; ++j) {
      const int c = tileN * 128 + wn * 32 + j * 16 + ln;
      const float bv = bias[c];
      for (int a = 0; a < 8; ++a) {
        const int r = tileM * 128 + wm * 64 + i * 16 + hi * 8 + a;
        const size_t off = (size_t)r * D_MODEL + c;
        out[off] = g.acc[i][j][a] + bv + resid[off];
      }
    }
  }
}

// ---------------------------------------------------------------------------
// Host entry.  Workspace layout (bytes):
//   h      bf16 [BT][D]        @ 0          ( 8 MB)
//   WqkvT  bf16 [3D][D]        @ 8388608    ( 6 MB)
//   q      bf16 [B,H,T,hd]     @ 14680064   ( 8 MB)
//   k      bf16 [B,H,T,hd]     @ 23068672   ( 8 MB)
//   vT     bf16 [B,H,hd,T]     @ 31457280   ( 8 MB)
//   WoutT  bf16 [D][D]         @ 39845888   ( 2 MB)
//   attn   bf16 [BT][D]        @ 41943040   ( 8 MB)   total ~50 MB
// ---------------------------------------------------------------------------
extern "C" void kernel_launch(void* const* d_in, const int* in_sizes, int n_in,
                              void* d_out, int out_size, void* d_ws, size_t ws_size,
                              hipStream_t stream) {
  const float* x      = (const float*)d_in[0];
  const float* W_qkv  = (const float*)d_in[1];
  const float* b_qkv  = (const float*)d_in[2];
  const float* W_out  = (const float*)d_in[3];
  const float* b_out  = (const float*)d_in[4];
  const float* gamma  = (const float*)d_in[5];
  const float* beta   = (const float*)d_in[6];
  float* out = (float*)d_out;

  char* ws = (char*)d_ws;
  unsigned short* h     = (unsigned short*)(ws);
  unsigned short* wqkvT = (unsigned short*)(ws + 8388608u);
  unsigned short* qb    = (unsigned short*)(ws + 14680064u);
  unsigned short* kb    = (unsigned short*)(ws + 23068672u);
  unsigned short* vT    = (unsigned short*)(ws + 31457280u);
  unsigned short* woutT = (unsigned short*)(ws + 39845888u);
  unsigned short* attn  = (unsigned short*)(ws + 41943040u);

  ln_bf16_kernel<<<BT, 256, 0, stream>>>(x, gamma, beta, h);
  transpose_f32_to_bf16<<<(D_MODEL * D3) / 256, 256, 0, stream>>>(
      W_qkv, wqkvT, D_MODEL, D3);
  transpose_f32_to_bf16<<<(D_MODEL * D_MODEL) / 256, 256, 0, stream>>>(
      W_out, woutT, D_MODEL, D_MODEL);
  qkv_gemm_kernel<<<dim3(D3 / 128, BT / 128), 256, 0, stream>>>(
      h, wqkvT, b_qkv, qb, kb, vT);
  attention_kernel<<<dim3(SEQ / 64, BATCH * N_HEADS), 128, 0, stream>>>(
      qb, kb, vT, attn);
  out_gemm_kernel<<<dim3(D_MODEL / 128, BT / 128), 256, 0, stream>>>(
      attn, woutT, b_out, x, out);
}
